// HierarchicalManifold_76965813944495
// MI455X (gfx1250) — matrix operand order repaired and checked by reading, not compiled, not run on hardware
//
#include <hip/hip_runtime.h>
#include <hip/hip_bf16.h>

// ---------------------------------------------------------------------------
// Types for CDNA5 WMMA (wave32): v_wmma_f32_16x16x32_bf16
// ---------------------------------------------------------------------------
typedef __bf16 bf16_t;
typedef __attribute__((ext_vector_type(8)))  __bf16 bf16x8_t;
typedef __attribute__((ext_vector_type(16))) __bf16 bf16x16_t;
typedef __attribute__((ext_vector_type(8)))  float  f32x8_t;

#define DEV_INLINE __device__ __forceinline__

DEV_INLINE f32x8_t f32x8_zero() {
  f32x8_t v = {0.f, 0.f, 0.f, 0.f, 0.f, 0.f, 0.f, 0.f};
  return v;
}

DEV_INLINE bf16x16_t cat8(bf16x8_t lo, bf16x8_t hi) {
  return __builtin_shufflevector(lo, hi, 0,1,2,3,4,5,6,7,8,9,10,11,12,13,14,15);
}

// A operand 16(M)x32(K) from row-major memory (lda in elements).
// ISA layout: lanes 0-15 -> M=lane, K chunks [0..7],[16..23];
//             lanes 16-31 -> M=lane-16, K chunks [8..15],[24..31].
DEV_INLINE bf16x16_t load_a_rm(const bf16_t* A, int lda, int lane) {
  int m  = lane & 15;
  int ko = (lane >> 4) << 3;          // 0 or 8
  const bf16_t* p = A + (long long)m * lda + ko;
  bf16x8_t lo = *(const bf16x8_t*)(p);
  bf16x8_t hi = *(const bf16x8_t*)(p + 16);
  return cat8(lo, hi);
}

// B operand 32(K)x16(N) from BT stored row-major as BT[n][k] (ldb in elements).
// ISA layout: lanes 0-15 -> col N=lane, K chunks [0..7],[16..23]; lanes 16-31 -> N=lane-16, K chunks [8..15],[24..31].
DEV_INLINE bf16x16_t load_b_ct(const bf16_t* BT, int ldb, int lane) {
  int n  = lane & 15;
  int ko = (lane >> 4) << 3;          // 0 or 8
  const bf16_t* p = BT + (long long)n * ldb + ko;
  bf16x8_t lo = *(const bf16x8_t*)(p);
  bf16x8_t hi = *(const bf16x8_t*)(p + 16);
  return cat8(lo, hi);
}

DEV_INLINE f32x8_t wmma_bf16(bf16x16_t a, bf16x16_t b, f32x8_t c) {
  // (neg_a, A, neg_b, B, c_mod, C, reuse_a, reuse_b)
  return __builtin_amdgcn_wmma_f32_16x16x32_bf16(false, a, false, b, (short)0, c,
                                                 false, false);
}
// C layout: VGPR r, lane l: M = (l>=16 ? 8 : 0) + r, N = l & 15.

// ---------------------------------------------------------------------------
// CDNA5 async global->LDS copy (tracked by ASYNCcnt).
// LDS byte offset is the low 32 bits of the generic pointer (aperture rule).
// ---------------------------------------------------------------------------
DEV_INLINE void async_load_b128(void* lds_dst, const void* gsrc) {
  unsigned int  loff = (unsigned int)(unsigned long long)(size_t)lds_dst;
  unsigned long long ga = (unsigned long long)(size_t)gsrc;
  asm volatile("global_load_async_to_lds_b128 %0, %1, off"
               :: "v"(loff), "v"(ga) : "memory");
}
DEV_INLINE void wait_asynccnt0() {
  asm volatile("s_wait_asynccnt 0" ::: "memory");
}

// ---------------------------------------------------------------------------
// Generic batched GEMM: C = act(alpha * A @ B + bias)
//   A: bf16 row-major [rows, K] per batch, BT: bf16 [ncols, K] (B transposed)
//   NT = 16-col tiles per wave (compile-time -> branch-free WMMA bursts).
//   Block = 256 threads (8 waves); wave handles 16 rows x NT*16 cols.
// ---------------------------------------------------------------------------
template <int NT>
__global__ void gemm_kernel(const bf16_t* __restrict__ A, int lda,
                            long long sAo, long long sAi,
                            const bf16_t* __restrict__ BT, int ldb,
                            long long sBo, long long sBi,
                            const float* __restrict__ bias,
                            float* __restrict__ Cf, bf16_t* __restrict__ Cb,
                            int ldc, long long sCo, long long sCi,
                            int inner, int rowsPerBatch, int K,
                            float alpha, int act, int transOut) {
  int rowTiles = (rowsPerBatch + 127) >> 7;
  int batch = blockIdx.x / rowTiles;
  int rt    = blockIdx.x % rowTiles;
  int bo = batch / inner, bi = batch % inner;
  const bf16_t* Ab  = A  + bo * sAo + bi * sAi;
  const bf16_t* BTb = BT + bo * sBo + bi * sBi;
  long long coff = (long long)bo * sCo + (long long)bi * sCi;

  int wave = threadIdx.x >> 5;
  int lane = threadIdx.x & 31;
  int row0 = rt * 128 + wave * 16;
  if (row0 >= rowsPerBatch) return;

  int colBase = blockIdx.y * 128;

  f32x8_t acc[NT];
#pragma unroll
  for (int t = 0; t < NT; t++) acc[t] = f32x8_zero();

  const bf16_t* Arow = Ab + (long long)row0 * lda;
  for (int kb = 0; kb < K; kb += 32) {
    bf16x16_t a = load_a_rm(Arow + kb, lda, lane);
    __builtin_prefetch(Arow + kb + 32, 0, 0);   // global_prefetch on gfx1250
    bf16x16_t bt[NT];
#pragma unroll
    for (int t = 0; t < NT; t++)
      bt[t] = load_b_ct(BTb + (long long)(colBase + t * 16) * ldb + kb, ldb, lane);
#pragma unroll
    for (int t = 0; t < NT; t++)
      acc[t] = wmma_bf16(a, bt[t], acc[t]);
  }

  int n0   = lane & 15;
  int moff = (lane >> 4) << 3;
#pragma unroll
  for (int t = 0; t < NT; t++) {
    int ncol = colBase + t * 16 + n0;
    float bv = bias ? bias[ncol] : 0.f;
#pragma unroll
    for (int r = 0; r < 8; r++) {
      float v = acc[t][r] * alpha + bv;
      if (act == 1) v = tanhf(v);
      int m = row0 + moff + r;
      if (Cf) Cf[coff + (long long)m * ldc + ncol] = v;
      if (Cb) {
        long long idx = transOut ? coff + (long long)ncol * ldc + m
                                 : coff + (long long)m * ldc + ncol;
        Cb[idx] = (bf16_t)v;
      }
    }
  }
}

// ---------------------------------------------------------------------------
// Fused automata step: h = tanh(S@W+b); mod = S@Tr; S' = S + 0.1*(h+mod)
//   Writes S' (fp32 + bf16) in place and h (bf16) for the trace update.
// ---------------------------------------------------------------------------
__global__ void automata_kernel(float* __restrict__ S, bf16_t* __restrict__ Sb,
                                const bf16_t* __restrict__ WT,
                                const float* __restrict__ bias,
                                const bf16_t* __restrict__ TrT,
                                bf16_t* __restrict__ Hb, int rowsPerBatch) {
  int rowTiles = (rowsPerBatch + 127) >> 7;
  int batch = blockIdx.x / rowTiles;
  int rt    = blockIdx.x % rowTiles;
  int wave = threadIdx.x >> 5;
  int lane = threadIdx.x & 31;
  int row0 = rt * 128 + wave * 16;
  if (row0 >= rowsPerBatch) return;

  long long sOff = (long long)batch * rowsPerBatch * 128;
  const bf16_t* Arow = Sb + sOff + (long long)row0 * 128;
  const bf16_t* trt  = TrT + (long long)batch * 16384;

  int n0   = lane & 15;
  int moff = (lane >> 4) << 3;

  // h = tanh(S @ W + b)
  f32x8_t acc[8];
#pragma unroll
  for (int t = 0; t < 8; t++) acc[t] = f32x8_zero();
  for (int kb = 0; kb < 128; kb += 32) {
    bf16x16_t a = load_a_rm(Arow + kb, 128, lane);
    bf16x16_t bt[8];
#pragma unroll
    for (int t = 0; t < 8; t++)
      bt[t] = load_b_ct(WT + (long long)(t * 16) * 128 + kb, 128, lane);
#pragma unroll
    for (int t = 0; t < 8; t++) acc[t] = wmma_bf16(a, bt[t], acc[t]);
  }
#pragma unroll
  for (int t = 0; t < 8; t++) {
    int n = t * 16 + n0;
    float bv = bias[n];
#pragma unroll
    for (int r = 0; r < 8; r++) {
      float h = tanhf(acc[t][r] + bv);
      acc[t][r] = h;                                       // keep h in regs
      int m = row0 + moff + r;
      Hb[sOff + (long long)m * 128 + n] = (bf16_t)h;
    }
  }

  // mod = S @ Tr (TrT is trace transposed, bf16)
  f32x8_t acc2[8];
#pragma unroll
  for (int t = 0; t < 8; t++) acc2[t] = f32x8_zero();
  for (int kb = 0; kb < 128; kb += 32) {
    bf16x16_t a = load_a_rm(Arow + kb, 128, lane);
    bf16x16_t bt[8];
#pragma unroll
    for (int t = 0; t < 8; t++)
      bt[t] = load_b_ct(trt + (long long)(t * 16) * 128 + kb, 128, lane);
#pragma unroll
    for (int t = 0; t < 8; t++) acc2[t] = wmma_bf16(a, bt[t], acc2[t]);
  }

  // S' = S + 0.1*(h + mod)
#pragma unroll
  for (int t = 0; t < 8; t++) {
    int n = t * 16 + n0;
#pragma unroll
    for (int r = 0; r < 8; r++) {
      int m = row0 + moff + r;
      long long idx = sOff + (long long)m * 128 + n;
      float nv = S[idx] + 0.1f * (acc[t][r] + acc2[t][r]);
      S[idx]  = nv;
      Sb[idx] = (bf16_t)nv;
    }
  }
}

// ---------------------------------------------------------------------------
// Hebbian trace update: Tr' = 0.9*Tr + scale * S^T @ H   (per-batch 128x128,
// K = rowsPerBatch). One block (8 waves) per batch.
// Double-buffered global_load_async_to_lds_b128 prefetches the next 32-row
// chunk of S/H while the current chunk is LDS-transposed and fed to WMMA.
// Also writes TrT (bf16, transposed) for next step's modulation GEMM.
// ---------------------------------------------------------------------------
__global__ void trace_kernel(float* __restrict__ Tr, bf16_t* __restrict__ TrT,
                             const bf16_t* __restrict__ Sb,
                             const bf16_t* __restrict__ Hb,
                             int Krows, float scale) {
  const int PAD = 40;                          // row stride (80B, 16B aligned)
  __shared__ bf16_t rawS[2][4096];             // 32 rows x 128 cols per chunk
  __shared__ bf16_t rawH[2][4096];
  __shared__ bf16_t ldsS[128 * PAD];           // transposed [d][k]
  __shared__ bf16_t ldsH[128 * PAD];

  int b = blockIdx.x;
  const bf16_t* S = Sb + (long long)b * Krows * 128;
  const bf16_t* H = Hb + (long long)b * Krows * 128;
  int wave = threadIdx.x >> 5;
  int lane = threadIdx.x & 31;
  int tid  = threadIdx.x;

  // prefetch chunk 0: each thread async-copies 2x16B per buffer
  {
    int e0 = tid * 8;            // 8 bf16 = 16B
    int e1 = e0 + 2048;
    async_load_b128(&rawS[0][e0], S + e0);
    async_load_b128(&rawH[0][e0], H + e0);
    async_load_b128(&rawS[0][e1], S + e1);
    async_load_b128(&rawH[0][e1], H + e1);
  }
  wait_asynccnt0();
  __syncthreads();

  f32x8_t acc[8];
#pragma unroll
  for (int t = 0; t < 8; t++) acc[t] = f32x8_zero();

  int nCh = Krows >> 5;
  for (int c = 0; c < nCh; c++) {
    int p = c & 1;
    if (c + 1 < nCh) {                          // async prefetch next chunk
      const bf16_t* Sn = S + (long long)(c + 1) * 4096;
      const bf16_t* Hn = H + (long long)(c + 1) * 4096;
      int e0 = tid * 8, e1 = e0 + 2048;
      async_load_b128(&rawS[1 - p][e0], Sn + e0);
      async_load_b128(&rawH[1 - p][e0], Hn + e0);
      async_load_b128(&rawS[1 - p][e1], Sn + e1);
      async_load_b128(&rawH[1 - p][e1], Hn + e1);
    }
    // LDS transpose raw[p] -> ldsS/ldsH
    for (int i = tid; i < 4096; i += 256) {
      int k = i >> 7;           // 0..31
      int d = i & 127;          // feature
      ldsS[d * PAD + k] = rawS[p][i];
      ldsH[d * PAD + k] = rawH[p][i];
    }
    __syncthreads();

    // A tile: rows d = wave*16..+15 of S^T (contiguous k in ldsS)
    {
      int m  = (lane & 15) + wave * 16;
      int ko = (lane >> 4) << 3;
      const bf16_t* pa = &ldsS[m * PAD + ko];
      bf16x8_t lo = *(const bf16x8_t*)(pa);
      bf16x8_t hi = *(const bf16x8_t*)(pa + 16);
      bf16x16_t a = cat8(lo, hi);
      bf16x16_t bt[8];
#pragma unroll
      for (int t = 0; t < 8; t++) {
        int n   = t * 16 + (lane & 15);
        int kob = (lane >> 4) << 3;
        const bf16_t* q = &ldsH[n * PAD + kob];
        bf16x8_t blo = *(const bf16x8_t*)(q);
        bf16x8_t bhi = *(const bf16x8_t*)(q + 16);
        bt[t] = cat8(blo, bhi);
      }
#pragma unroll
      for (int t = 0; t < 8; t++) acc[t] = wmma_bf16(a, bt[t], acc[t]);
    }
    wait_asynccnt0();          // next chunk's async copies landed
    __syncthreads();           // and everyone is past this chunk's reads
  }

  int n0   = lane & 15;
  int moff = (lane >> 4) << 3;
  long long bOff = (long long)b * 16384;
#pragma unroll
  for (int t = 0; t < 8; t++) {
    int n = t * 16 + n0;
#pragma unroll
    for (int r = 0; r < 8; r++) {
      int m = wave * 16 + moff + r;
      long long idx = bOff + (long long)m * 128 + n;
      float nv = 0.9f * Tr[idx] + scale * acc[t][r];
      Tr[idx] = nv;
      TrT[bOff + (long long)n * 128 + m] = (bf16_t)nv;   // transposed bf16
    }
  }
}

// ---------------------------------------------------------------------------
// Small helper kernels
// ---------------------------------------------------------------------------
__global__ void wtrans_kernel(const float* __restrict__ W, bf16_t* __restrict__ WT) {
  for (int i = blockIdx.x * blockDim.x + threadIdx.x; i < 16384;
       i += gridDim.x * blockDim.x) {
    int n = i >> 7, k = i & 127;
    WT[i] = (bf16_t)W[k * 128 + n];     // WT[n][k] = W[k][n]
  }
}

__global__ void init_kernel(const float* __restrict__ microIn,
                            float* __restrict__ micro,
                            float* __restrict__ macro, bf16_t* __restrict__ macrob,
                            float* __restrict__ mtr, float* __restrict__ Mtr,
                            bf16_t* __restrict__ mtrT, bf16_t* __restrict__ MtrT,
                            const float* __restrict__ vid) {
  long long stride = (long long)gridDim.x * blockDim.x;
  for (long long i = (long long)blockIdx.x * blockDim.x + threadIdx.x;
       i < 33554432LL; i += stride) {
    micro[i] = microIn[i];
    if (i < 2097152LL) {
      float v = vid[i & 127];
      macro[i] = v;
      macrob[i] = (bf16_t)v;
    }
    if (i < 4194304LL) {
      mtr[i] = 0.f; Mtr[i] = 0.f;
      mtrT[i] = (bf16_t)0.f; MtrT[i] = (bf16_t)0.f;
    }
  }
}

// bias = mean(macro over 64 nodes) @ W_td + b_td  -> [B,128]
__global__ void bias_kernel(const float* __restrict__ macro,
                            const float* __restrict__ W_td,
                            const float* __restrict__ b_td,
                            float* __restrict__ biasOut) {
  int b = blockIdx.x;
  int d = threadIdx.x;            // 128 threads
  const float* M = macro + (long long)b * 64 * 128;
  float s = 0.f;
  for (int n = 0; n < 64; n++) s += M[n * 128 + d];
  __shared__ float mm[128];
  mm[d] = s * (1.f / 64.f);
  __syncthreads();
  float acc = b_td[d];
  for (int e = 0; e < 128; e++) acc += mm[e] * W_td[e * 128 + d];
  biasOut[b * 128 + d] = acc;
}

// micro += 0.1*bias (broadcast over nodes); mirror to bf16
__global__ void micro_pre_kernel(float* __restrict__ micro,
                                 bf16_t* __restrict__ microb,
                                 const float* __restrict__ bias) {
  long long stride = (long long)gridDim.x * blockDim.x;
  for (long long i = (long long)blockIdx.x * blockDim.x + threadIdx.x;
       i < 33554432LL; i += stride) {
    int d = (int)(i & 127);
    long long b = i >> 17;                   // 1024*128 = 2^17
    float v = micro[i] + 0.1f * bias[b * 128 + d];
    micro[i] = v;
    microb[i] = (bf16_t)v;
  }
}

__global__ void softmax_kernel(const float* __restrict__ S, bf16_t* __restrict__ A) {
  long long row = blockIdx.x;
  const float* s = S + row * 1024;
  bf16_t* o = A + row * 1024;
  int tid = threadIdx.x;          // 256
  float v[4];
  float mx = -3.4e38f;
#pragma unroll
  for (int j = 0; j < 4; j++) { v[j] = s[tid + j * 256]; mx = fmaxf(mx, v[j]); }
  __shared__ float red[256];
  red[tid] = mx; __syncthreads();
  for (int st = 128; st > 0; st >>= 1) {
    if (tid < st) red[tid] = fmaxf(red[tid], red[tid + st]);
    __syncthreads();
  }
  mx = red[0]; __syncthreads();
  float sum = 0.f;
#pragma unroll
  for (int j = 0; j < 4; j++) { v[j] = __expf(v[j] - mx); sum += v[j]; }
  red[tid] = sum; __syncthreads();
  for (int st = 128; st > 0; st >>= 1) {
    if (tid < st) red[tid] += red[tid + st];
    __syncthreads();
  }
  float inv = 1.f / red[0];
#pragma unroll
  for (int j = 0; j < 4; j++) o[tid + j * 256] = (bf16_t)(v[j] * inv);
}

// macro = LN(macro + 0.1*integ); mirror bf16. One block (128 thr) per row.
__global__ void ln_kernel(float* __restrict__ macro, bf16_t* __restrict__ macrob,
                          const float* __restrict__ integ,
                          const float* __restrict__ g, const float* __restrict__ bb) {
  long long row = blockIdx.x;
  int d = threadIdx.x;            // 128
  long long idx = row * 128 + d;
  float x = macro[idx] + 0.1f * integ[idx];
  __shared__ float red[128];
  red[d] = x; __syncthreads();
  for (int st = 64; st > 0; st >>= 1) {
    if (d < st) red[d] += red[d + st];
    __syncthreads();
  }
  float mean = red[0] * (1.f / 128.f);
  __syncthreads();
  float dx = x - mean;
  red[d] = dx * dx; __syncthreads();
  for (int st = 64; st > 0; st >>= 1) {
    if (d < st) red[d] += red[d + st];
    __syncthreads();
  }
  float var = red[0] * (1.f / 128.f);
  float y = dx * rsqrtf(var + 1e-5f) * g[d] + bb[d];
  macro[idx] = y;
  macrob[idx] = (bf16_t)y;
}

__global__ void blend_kernel(float* __restrict__ macro, bf16_t* __restrict__ macrob,
                             const float* __restrict__ vid) {
  long long stride = (long long)gridDim.x * blockDim.x;
  for (long long i = (long long)blockIdx.x * blockDim.x + threadIdx.x;
       i < 2097152LL; i += stride) {
    float v = 0.95f * macro[i] + 0.05f * vid[i & 127];
    macro[i] = v;
    macrob[i] = (bf16_t)v;
  }
}

__global__ void energy_init_kernel(float* __restrict__ e,
                                   const float* __restrict__ b_e2) {
  if (threadIdx.x < 256) e[threadIdx.x] = b_e2[0];
}

// energy[b] += (1/1024) * sum_n ( tanh(micro@W_e1+b_e1) . W_e2 )
__global__ void energy_kernel(const bf16_t* __restrict__ microb,
                              const bf16_t* __restrict__ We1T,
                              const float* __restrict__ b_e1,
                              const float* __restrict__ We2,
                              float* __restrict__ energy) {
  int b  = blockIdx.x >> 3;
  int rt = blockIdx.x & 7;
  int wave = threadIdx.x >> 5;
  int lane = threadIdx.x & 31;
  int row0 = rt * 128 + wave * 16;
  const bf16_t* Arow = microb + (long long)b * 131072 + (long long)row0 * 128;

  f32x8_t acc[8];
#pragma unroll
  for (int t = 0; t < 8; t++) acc[t] = f32x8_zero();
  for (int kb = 0; kb < 128; kb += 32) {
    bf16x16_t a = load_a_rm(Arow + kb, 128, lane);
    bf16x16_t bt[8];
#pragma unroll
    for (int t = 0; t < 8; t++)
      bt[t] = load_b_ct(We1T + (long long)(t * 16) * 128 + kb, 128, lane);
#pragma unroll
    for (int t = 0; t < 8; t++) acc[t] = wmma_bf16(a, bt[t], acc[t]);
  }
  int n0 = lane & 15;
  float partial = 0.f;
#pragma unroll
  for (int t = 0; t < 8; t++) {
    int n = t * 16 + n0;
    float bv = b_e1[n], w2 = We2[n];
#pragma unroll
    for (int r = 0; r < 8; r++) partial += tanhf(acc[t][r] + bv) * w2;
  }
  for (int off = 16; off > 0; off >>= 1) partial += __shfl_xor(partial, off, 32);
  if (lane == 0) atomicAdd(&energy[b], partial * (1.f / 1024.f));
}

// ---------------------------------------------------------------------------
// Host side
// ---------------------------------------------------------------------------
static char* wbump(char*& p, size_t bytes) {
  char* r = p;
  p += (bytes + 255) & ~(size_t)255;
  return r;
}

extern "C" void kernel_launch(void* const* d_in, const int* in_sizes, int n_in,
                              void* d_out, int out_size, void* d_ws, size_t ws_size,
                              hipStream_t stream) {
  const int B = 256, NMI = 1024, NMA = 64, D = 128, STEPS = 5;  // steps fixed by setup
  (void)in_sizes; (void)n_in; (void)out_size; (void)ws_size;

  const float* microIn = (const float*)d_in[0];
  const float* W_td = (const float*)d_in[1];
  const float* b_td = (const float*)d_in[2];
  const float* ln_g = (const float*)d_in[3];
  const float* ln_b = (const float*)d_in[4];
  const float* W_q = (const float*)d_in[5];
  const float* b_q = (const float*)d_in[6];
  const float* W_k = (const float*)d_in[7];
  const float* b_k = (const float*)d_in[8];
  const float* W_v = (const float*)d_in[9];
  const float* b_v = (const float*)d_in[10];
  const float* W_o = (const float*)d_in[11];
  const float* b_o = (const float*)d_in[12];
  const float* W_mi = (const float*)d_in[13];
  const float* b_mi = (const float*)d_in[14];
  const float* W_ma = (const float*)d_in[15];
  const float* b_ma = (const float*)d_in[16];
  const float* W_e1 = (const float*)d_in[17];
  const float* b_e1 = (const float*)d_in[18];
  const float* W_e2 = (const float*)d_in[19];
  const float* b_e2 = (const float*)d_in[20];
  const float* v_id = (const float*)d_in[21];

  // Output regions (fp32 working state lives directly in d_out)
  float* out_micro  = (float*)d_out;
  float* out_macro  = out_micro + (size_t)B * NMI * D;
  float* out_mtr    = out_macro + (size_t)B * NMA * D;
  float* out_Mtr    = out_mtr + (size_t)B * D * D;
  float* out_energy = out_Mtr + (size_t)B * D * D;

  // Workspace carve
  char* wp = (char*)d_ws;
  bf16_t* WT_mi = (bf16_t*)wbump(wp, 16384 * 2);
  bf16_t* WT_q  = (bf16_t*)wbump(wp, 16384 * 2);
  bf16_t* WT_k  = (bf16_t*)wbump(wp, 16384 * 2);
  bf16_t* WT_v  = (bf16_t*)wbump(wp, 16384 * 2);
  bf16_t* WT_o  = (bf16_t*)wbump(wp, 16384 * 2);
  bf16_t* WT_ma = (bf16_t*)wbump(wp, 16384 * 2);
  bf16_t* WT_e1 = (bf16_t*)wbump(wp, 16384 * 2);
  bf16_t* micro_bf = (bf16_t*)wbump(wp, (size_t)B * NMI * D * 2);
  bf16_t* h_bf     = (bf16_t*)wbump(wp, (size_t)B * NMI * D * 2);
  bf16_t* kh_bf    = (bf16_t*)wbump(wp, (size_t)B * NMI * D * 2);
  bf16_t* vhT_bf   = (bf16_t*)wbump(wp, (size_t)B * D * NMI * 2);     // [B,128,1024]
  bf16_t* attn_bf  = (bf16_t*)wbump(wp, (size_t)B * 2 * 64 * 1024 * 2);
  float*  scores   = (float*) wbump(wp, (size_t)B * 2 * 64 * 1024 * 4);
  bf16_t* qh_bf    = (bf16_t*)wbump(wp, (size_t)B * NMA * D * 2);
  bf16_t* attout_bf= (bf16_t*)wbump(wp, (size_t)B * NMA * D * 2);
  bf16_t* h2_bf    = (bf16_t*)wbump(wp, (size_t)B * NMA * D * 2);
  bf16_t* macro_bf = (bf16_t*)wbump(wp, (size_t)B * NMA * D * 2);
  bf16_t* mtrT     = (bf16_t*)wbump(wp, (size_t)B * D * D * 2);
  bf16_t* MtrT     = (bf16_t*)wbump(wp, (size_t)B * D * D * 2);
  float*  integ    = (float*) wbump(wp, (size_t)B * NMA * D * 4);
  float*  biasws   = (float*) wbump(wp, (size_t)B * D * 4);

  // One-time prep
  wtrans_kernel<<<64, 256, 0, stream>>>(W_mi, WT_mi);
  wtrans_kernel<<<64, 256, 0, stream>>>(W_q,  WT_q);
  wtrans_kernel<<<64, 256, 0, stream>>>(W_k,  WT_k);
  wtrans_kernel<<<64, 256, 0, stream>>>(W_v,  WT_v);
  wtrans_kernel<<<64, 256, 0, stream>>>(W_o,  WT_o);
  wtrans_kernel<<<64, 256, 0, stream>>>(W_ma, WT_ma);
  wtrans_kernel<<<64, 256, 0, stream>>>(W_e1, WT_e1);
  init_kernel<<<4096, 256, 0, stream>>>(microIn, out_micro, out_macro, macro_bf,
                                        out_mtr, out_Mtr, mtrT, MtrT, v_id);

  const long long mS = (long long)NMI * D;          // 131072 micro batch stride
  const long long aS = (long long)NMA * D;          // 8192 macro batch stride

  for (int step = 0; step < STEPS; step++) {
    // top-down bias into micro
    bias_kernel<<<B, 128, 0, stream>>>(out_macro, W_td, b_td, biasws);
    micro_pre_kernel<<<4096, 256, 0, stream>>>(out_micro, micro_bf, biasws);

    // micro automata + trace
    automata_kernel<<<B * 8, 256, 0, stream>>>(out_micro, micro_bf, WT_mi, b_mi,
                                               mtrT, h_bf, NMI);
    trace_kernel<<<B, 256, 0, stream>>>(out_mtr, mtrT, micro_bf, h_bf, NMI,
                                        0.1f / (float)NMI);

    // attention projections
    gemm_kernel<8><<<dim3(B * 8, 1), 256, 0, stream>>>(
        micro_bf, 128, mS, 0, WT_k, 128, 0, 0, b_k,
        nullptr, kh_bf, 128, mS, 0, 1, NMI, 128, 1.f, 0, 0);
    gemm_kernel<8><<<dim3(B * 8, 1), 256, 0, stream>>>(
        micro_bf, 128, mS, 0, WT_v, 128, 0, 0, b_v,
        nullptr, vhT_bf, 1024, mS, 0, 1, NMI, 128, 1.f, 0, 1);  // transposed out
    gemm_kernel<8><<<dim3(B, 1), 256, 0, stream>>>(
        macro_bf, 128, aS, 0, WT_q, 128, 0, 0, b_q,
        nullptr, qh_bf, 128, aS, 0, 1, NMA, 128, 1.f, 0, 0);

    // scores[b,h] = (Q_h @ K_h^T) / 8 ; BT == kh row-major per head
    gemm_kernel<8><<<dim3(B * 2, 8), 256, 0, stream>>>(
        qh_bf, 128, aS, 64, kh_bf, 128, mS, 64, nullptr,
        scores, nullptr, 1024, (long long)2 * 64 * 1024, (long long)64 * 1024,
        2, 64, 64, 0.125f, 0, 0);
    softmax_kernel<<<B * 2 * 64, 256, 0, stream>>>(scores, attn_bf);

    // out[b,h] = attn @ V_h  (V supplied transposed: vhT[b, hd, n])
    gemm_kernel<4><<<dim3(B * 2, 1), 256, 0, stream>>>(
        attn_bf, 1024, (long long)2 * 64 * 1024, (long long)64 * 1024,
        vhT_bf, 1024, (long long)D * NMI, (long long)64 * 1024, nullptr,
        nullptr, attout_bf, 128, aS, 64, 2, 64, 1024, 1.f, 0, 0);

    // integ = out @ W_o + b_o
    gemm_kernel<8><<<dim3(B, 1), 256, 0, stream>>>(
        attout_bf, 128, aS, 0, WT_o, 128, 0, 0, b_o,
        integ, nullptr, 128, aS, 0, 1, NMA, 128, 1.f, 0, 0);

    // macro = LN(macro + 0.1*integ); macro automata + trace; identity blend
    ln_kernel<<<B * NMA, 128, 0, stream>>>(out_macro, macro_bf, integ, ln_g, ln_b);
    automata_kernel<<<B, 256, 0, stream>>>(out_macro, macro_bf, WT_ma, b_ma,
                                           MtrT, h2_bf, NMA);
    trace_kernel<<<B, 256, 0, stream>>>(out_Mtr, MtrT, macro_bf, h2_bf, NMA,
                                        0.1f / (float)NMA);
    blend_kernel<<<1024, 256, 0, stream>>>(out_macro, macro_bf, v_id);
  }

  // energy
  energy_init_kernel<<<1, 256, 0, stream>>>(out_energy, b_e2);
  energy_kernel<<<B * 8, 256, 0, stream>>>(micro_bf, WT_e1, b_e1, W_e2, out_energy);
}
